// Sequence_23106924052662
// MI455X (gfx1250) — matrix operand order repaired and checked by reading, not compiled, not run on hardware
//
#include <hip/hip_runtime.h>

typedef _Float16 half_t;
typedef _Float16 v16h __attribute__((ext_vector_type(16)));
typedef _Float16 v8h  __attribute__((ext_vector_type(8)));
typedef float    v8f  __attribute__((ext_vector_type(8)));

#define LSTM_L   100
#define NPAD     112      // 7 * 16  (per-gate N, padded)
#define KPAD     128      // 4 * 32  (K, padded)
#define ROWS     16       // batch rows per workgroup (one WMMA M-tile)
#define NTILES   7
#define KSTEPS   4
#define THREADS  128      // 4 waves, one per gate

__device__ __forceinline__ float sigmoidf_fast(float x) {
    return 1.0f / (1.0f + __expf(-x));
}
__device__ __forceinline__ float tanhf_fast(float x) {
    x = fminf(fmaxf(x, -15.0f), 15.0f);
    float e = __expf(2.0f * x);
    return (e - 1.0f) / (e + 1.0f);
}

union Frag16 { v16h v; v8h h[2]; };

__global__ void __launch_bounds__(THREADS, 1) lstm2layer_wmma_kernel(
    const float* __restrict__ x,      // [B, T]
    const float* __restrict__ Wih1,   // [400, 1]
    const float* __restrict__ Whh1,   // [400, 100]
    const float* __restrict__ bih1,   // [400]
    const float* __restrict__ bhh1,   // [400]
    const float* __restrict__ Wih2,   // [4, 100]
    const float* __restrict__ Whh2,   // [4, 1]
    const float* __restrict__ bih2,   // [4]
    const float* __restrict__ bhh2,   // [4]
    float* __restrict__ out,          // [B, T]
    int T)
{
    // ---- static LDS (~45 KB) ----
    __shared__ half_t h1l[ROWS * KPAD];        // f16 hidden state, K zero-padded
    __shared__ float  c1l[ROWS * NPAD];        // f32 cell state
    __shared__ float  Gl[4 * ROWS * NPAD];     // gate pre-activations from WMMA
    __shared__ float  b1l[4 * LSTM_L];         // bih1 + bhh1
    __shared__ float  wxl[4 * LSTM_L];         // W_ih1 column
    __shared__ float  w2l[4 * LSTM_L];         // W_ih2 row-major
    __shared__ float  whh2l[4];
    __shared__ float  b2l[4];
    __shared__ float  xv[ROWS];
    __shared__ float  h2l[ROWS];
    __shared__ float  c2l[ROWS];
    __shared__ float  g2p[ROWS * 4 * 2];       // LSTM2 dot partials

    const int tid  = threadIdx.x;
    const int wave = tid >> 5;        // gate index 0..3 (i,f,g,o)
    const int lane = tid & 31;
    const int hi   = (lane >= 16) ? 1 : 0;
    const int nl   = lane & 15;
    const int row0 = blockIdx.x * ROWS;

    // ---- one-time LDS init ----
    for (int idx = tid; idx < ROWS * KPAD; idx += THREADS) h1l[idx] = (half_t)0.0f;
    for (int idx = tid; idx < ROWS * NPAD; idx += THREADS) c1l[idx] = 0.0f;
    for (int idx = tid; idx < 4 * LSTM_L; idx += THREADS) {
        b1l[idx] = bih1[idx] + bhh1[idx];
        wxl[idx] = Wih1[idx];
        w2l[idx] = Wih2[idx];
    }
    if (tid < 4) { whh2l[tid] = Whh2[tid]; b2l[tid] = bih2[tid] + bhh2[tid]; }
    if (tid < ROWS) { h2l[tid] = 0.0f; c2l[tid] = 0.0f; }

    // ---- load ALL recurrent-weight B-fragments into registers (loop invariant).
    // B 32x16 f16 layout: lane<16 -> col=lane,   K = ks*32 + 0..15
    //                     lane>=16 -> col=lane-16, K = ks*32 + 16..31
    v16h bfr[KSTEPS][NTILES];
    #pragma unroll
    for (int ks = 0; ks < KSTEPS; ++ks) {
        #pragma unroll
        for (int nt = 0; nt < NTILES; ++nt) {
            v16h tmp;
            const int cc = nt * 16 + nl;                 // output-neuron index within gate
            #pragma unroll
            for (int i = 0; i < 16; ++i) {
                const int kk  = ks * 32 + hi * 16 + i;   // hidden index
                const bool ok = (cc < LSTM_L) && (kk < LSTM_L);
                const int idx = ok ? ((wave * LSTM_L + cc) * LSTM_L + kk) : 0;
                float w = Whh1[idx];
                tmp[i] = (half_t)(ok ? w : 0.0f);
            }
            bfr[ks][nt] = tmp;
        }
    }
    __syncthreads();

    for (int t = 0; t < T; ++t) {
        // ---- phase 0: stage x_t for this row tile ----
        if (tid < ROWS) {
            const float* px = x + (size_t)(row0 + tid) * T + t;
            xv[tid] = *px;
            __builtin_prefetch(px + 8, 0, 0);   // global_prefetch_b8 ahead in the x stream
        }
        __syncthreads();

        // ---- phase 1: WMMA gates = h1 @ W_hh1^T (each wave owns one gate) ----
        {
            // preload all four A fragments (8 x ds_load_b128), then pure WMMA burst
            Frag16 a[KSTEPS];
            #pragma unroll
            for (int ks = 0; ks < KSTEPS; ++ks) {
                const half_t* ap = h1l + nl * KPAD + ks * 32 + hi * 16;
                a[ks].h[0] = *(const v8h*)(ap);
                a[ks].h[1] = *(const v8h*)(ap + 8);
            }
            v8f acc[NTILES];
            #pragma unroll
            for (int nt = 0; nt < NTILES; ++nt) acc[nt] = (v8f){};
            #pragma unroll
            for (int ks = 0; ks < KSTEPS; ++ks) {
                #pragma unroll
                for (int nt = 0; nt < NTILES; ++nt) {
                    acc[nt] = __builtin_amdgcn_wmma_f32_16x16x32_f16(
                        false, a[ks].v, false, bfr[ks][nt], (short)0, acc[nt], false, false);
                }
            }
            // scatter C tiles: VGPR r, lanes 0-15 -> M=r, N=lane; lanes 16-31 -> M=r+8, N=lane-16
            #pragma unroll
            for (int nt = 0; nt < NTILES; ++nt) {
                #pragma unroll
                for (int r = 0; r < 8; ++r) {
                    Gl[(wave * ROWS + r + hi * 8) * NPAD + nt * 16 + nl] = acc[nt][r];
                }
            }
        }
        __syncthreads();

        // ---- phase 2: elementwise LSTM1 update (adds x*W_ih1 + bias rank-1 term) ----
        for (int e = tid; e < ROWS * LSTM_L; e += THREADS) {
            int row = e / LSTM_L;
            int n   = e % LSTM_L;
            float xr = xv[row];
            float gi = Gl[(0 * ROWS + row) * NPAD + n] + xr * wxl[0 * LSTM_L + n] + b1l[0 * LSTM_L + n];
            float gf = Gl[(1 * ROWS + row) * NPAD + n] + xr * wxl[1 * LSTM_L + n] + b1l[1 * LSTM_L + n];
            float gg = Gl[(2 * ROWS + row) * NPAD + n] + xr * wxl[2 * LSTM_L + n] + b1l[2 * LSTM_L + n];
            float go = Gl[(3 * ROWS + row) * NPAD + n] + xr * wxl[3 * LSTM_L + n] + b1l[3 * LSTM_L + n];
            float i_ = sigmoidf_fast(gi);
            float f_ = sigmoidf_fast(gf);
            float g_ = tanhf_fast(gg);
            float o_ = sigmoidf_fast(go);
            float cn = f_ * c1l[row * NPAD + n] + i_ * g_;
            float hn = o_ * tanhf_fast(cn);
            c1l[row * NPAD + n] = cn;
            h1l[row * KPAD + n] = (half_t)hn;   // K >= 100 stays zero from init
        }
        __syncthreads();

        // ---- phase 3: LSTM2 gate dots, 2 threads per (row, gate) dot ----
        {
            int row = tid >> 3;          // 0..15
            int j   = (tid >> 1) & 3;    // gate
            int hv  = tid & 1;           // which half of K
            float s = 0.0f;
            const half_t* hp = h1l + row * KPAD + hv * 50;
            const float*  wp = w2l + j * LSTM_L + hv * 50;
            #pragma unroll 10
            for (int k = 0; k < 50; ++k) s += (float)hp[k] * wp[k];
            g2p[tid] = s;   // [row][j][hv]
        }
        __syncthreads();

        // ---- phase 4: LSTM2 state update + emit output ----
        if (tid < ROWS) {
            int row = tid;
            const float* gp = g2p + row * 8;
            float pi = gp[0] + gp[1] + b2l[0] + whh2l[0] * h2l[row];
            float pf = gp[2] + gp[3] + b2l[1] + whh2l[1] * h2l[row];
            float pg = gp[4] + gp[5] + b2l[2] + whh2l[2] * h2l[row];
            float po = gp[6] + gp[7] + b2l[3] + whh2l[3] * h2l[row];
            float i_ = sigmoidf_fast(pi);
            float f_ = sigmoidf_fast(pf);
            float g_ = tanhf_fast(pg);
            float o_ = sigmoidf_fast(po);
            float cn = f_ * c2l[row] + i_ * g_;
            float hn = o_ * tanhf_fast(cn);
            c2l[row] = cn;
            h2l[row] = hn;
            out[(size_t)(row0 + row) * T + t] = hn;
        }
        __syncthreads();   // h2/c2 updates visible before next step's phase 3/4
    }
}

extern "C" void kernel_launch(void* const* d_in, const int* in_sizes, int n_in,
                              void* d_out, int out_size, void* d_ws, size_t ws_size,
                              hipStream_t stream) {
    const float* x    = (const float*)d_in[0];
    const float* Wih1 = (const float*)d_in[1];
    const float* Whh1 = (const float*)d_in[2];
    const float* bih1 = (const float*)d_in[3];
    const float* bhh1 = (const float*)d_in[4];
    const float* Wih2 = (const float*)d_in[5];
    const float* Whh2 = (const float*)d_in[6];
    const float* bih2 = (const float*)d_in[7];
    const float* bhh2 = (const float*)d_in[8];
    float* out = (float*)d_out;

    const int T = 2048;                 // reference hardcodes T
    const int B = in_sizes[0] / T;      // 4096

    dim3 grid(B / ROWS);
    dim3 block(THREADS);
    lstm2layer_wmma_kernel<<<grid, block, 0, stream>>>(
        x, Wih1, Whh1, bih1, bhh1, Wih2, Whh2, bih2, bhh2, out, T);
}
